// Eagle3LlamaAttention_26517128086127
// MI455X (gfx1250) — matrix operand order repaired and checked by reading, not compiled, not run on hardware
//
#include <hip/hip_runtime.h>
#include <hip/hip_bf16.h>

// ---------------------------------------------------------------------------
// Eagle3 Llama attention for MI455X (gfx1250), wave32 + WMMA bf16 path.
//   B=1, S=2048, H=32, KVH=8, D=128, HID=4096
// ---------------------------------------------------------------------------

#define S_LEN   2048
#define NHEADS  32
#define NKVH    8
#define HDIM    128
#define QSTRIDE 4096      // H*D
#define KSTRIDE 1024      // KVH*D
#define XK      8192      // 2*HID
#define SCALE_F 0.08838834764831845f   // 128^-0.5
#define NEG_F   (-1.0e9f)
#define LCK_I   2
#define LOG_THETA 9.210340371976184f   // ln(10000)

typedef __bf16 bf16_t;
typedef __attribute__((ext_vector_type(16))) bf16_t v16bf;
typedef __attribute__((ext_vector_type(8)))  float  v8f;
typedef __attribute__((ext_vector_type(4)))  int    v4i_t;

union FragU { v16bf v; uint4 q[2]; };

// ---- CDNA5 async global->LDS path (guarded; falls back to manual loads) ----
#if __has_builtin(__builtin_amdgcn_global_load_async_to_lds_b128)
#define ASYNC_LDS_OK 1
__device__ __forceinline__ void async_copy_b128(const void* gptr, void* lptr) {
    // prototype (from hipcc diagnostic): (v4i*, v4i*, imm offset, imm cpol)
    __builtin_amdgcn_global_load_async_to_lds_b128((v4i_t*)gptr, (v4i_t*)lptr, 0, 0);
}
#endif

__device__ __forceinline__ void wait_async0() {
#if __has_builtin(__builtin_amdgcn_s_wait_asynccnt)
    __builtin_amdgcn_s_wait_asynccnt(0);
#else
    asm volatile("s_wait_asynccnt 0" ::: "memory");
#endif
}

__device__ __forceinline__ float bfbits2f(unsigned short v) {
    union { unsigned u; float f; } a; a.u = ((unsigned)v) << 16; return a.f;
}
__device__ __forceinline__ unsigned short f2bfbits(float f) {
    union { float f; unsigned u; } a; a.f = f;
    unsigned r = a.u + 0x7FFFu + ((a.u >> 16) & 1u);
    return (unsigned short)(r >> 16);
}

__device__ __forceinline__ v8f wmma_bf16(const FragU& a, const FragU& b, v8f c) {
    return __builtin_amdgcn_wmma_f32_16x16x32_bf16(
        /*neg_a=*/false, a.v, /*neg_b=*/false, b.v,
        /*c_mod=*/(short)0, c, /*reuse_a=*/false, /*reuse_b=*/false);
}

// ---------------------------------------------------------------------------
// 1) fp32 -> bf16 streaming conversion (one HBM pass per operand)
// ---------------------------------------------------------------------------
__global__ __launch_bounds__(256) void cvt_f32_bf16(
    const float* __restrict__ src, unsigned short* __restrict__ dst, long long n)
{
    long long i = ((long long)blockIdx.x * 256 + threadIdx.x) * 4;
    if (i + 3 < n) {
        float4 v = *(const float4*)(src + i);
        ushort4 o;
        o.x = f2bfbits(v.x); o.y = f2bfbits(v.y);
        o.z = f2bfbits(v.z); o.w = f2bfbits(v.w);
        *(ushort4*)(dst + i) = o;
    } else {
        for (; i < n; ++i) dst[i] = f2bfbits(src[i]);
    }
}

// ---------------------------------------------------------------------------
// 2) bf16 GEMM:  C[M,N] = A[M,K] * B[K,N]
//    128x128 macrotile, 256 threads = 8 waves in a 2(m) x 4(n) grid,
//    each wave computes 64x32 via 4x2 WMMA 16x16 tiles, K-step 32.
//    A tile staged via GLOBAL_LOAD_ASYNC_TO_LDS (ASYNCcnt), B transposed manually.
// ---------------------------------------------------------------------------
template <bool OUT_F32>
__global__ __launch_bounds__(256) void gemm_bf16(
    const unsigned short* __restrict__ A,
    const unsigned short* __restrict__ B,
    void* __restrict__ C, int M, int N, int K)
{
    __shared__ unsigned short ldsA[128][40];    // [m][k], padded
    __shared__ unsigned short ldsBt[128][40];   // [n][k], transposed, padded

    const int tid  = threadIdx.x;
    const int lane = tid & 31, w = tid >> 5;
    const int wm = w >> 2, wn = w & 3;
    const int g = lane >> 4, l15 = lane & 15;
    const int m0 = blockIdx.y * 128, n0 = blockIdx.x * 128;

    v8f acc[4][2] = {};

    for (int k0 = 0; k0 < K; k0 += 32) {
        __syncthreads();
        // A tile 128x32 : 512 x b128 transfers, async global->LDS when available
        for (int i = tid; i < 512; i += 256) {
            int row = i >> 2, c4 = (i & 3) * 8;
            const unsigned short* gp = A + (size_t)(m0 + row) * K + k0 + c4;
#ifdef ASYNC_LDS_OK
            async_copy_b128(gp, &ldsA[row][c4]);
#else
            *(uint4*)&ldsA[row][c4] = *(const uint4*)gp;
#endif
        }
        // B tile 32x128 -> transposed LDS (manual: async copies cannot transpose)
        for (int i = tid; i < 512; i += 256) {
            int kr = i >> 4, nb = (i & 15) * 8;
            uint4 v = *(const uint4*)(B + (size_t)(k0 + kr) * N + n0 + nb);
            const unsigned short* e = (const unsigned short*)&v;
            #pragma unroll
            for (int j = 0; j < 8; ++j) ldsBt[nb + j][kr] = e[j];
        }
#ifdef ASYNC_LDS_OK
        wait_async0();
#endif
        __syncthreads();

        if (k0 + 32 < K) {
            __builtin_prefetch(A + (size_t)(m0 + (tid >> 1)) * K + k0 + 32, 0, 1);
            __builtin_prefetch(B + (size_t)(k0 + 32 + (tid & 31)) * N + n0, 0, 1);
        }

        // A fragments (ISA 16-bit A layout: lane=M, K-halves split by lane group)
        FragU a[4];
        #pragma unroll
        for (int tm = 0; tm < 4; ++tm) {
            int row = wm * 64 + tm * 16 + l15;
            int kb = g * 8;
            a[tm].q[0] = *(const uint4*)&ldsA[row][kb];
            a[tm].q[1] = *(const uint4*)&ldsA[row][kb + 16];
        }
        #pragma unroll
        for (int tn = 0; tn < 2; ++tn) {
            FragU b;
            int col = wn * 32 + tn * 16 + l15;
            int kb = g * 8;
            b.q[0] = *(const uint4*)&ldsBt[col][kb];
            b.q[1] = *(const uint4*)&ldsBt[col][kb + 16];
            #pragma unroll
            for (int tm = 0; tm < 4; ++tm)
                acc[tm][tn] = wmma_bf16(a[tm], b, acc[tm][tn]);
        }
    }

    // epilogue (C layout: vgpr r -> M = r + 8*(lane/16), N = lane%16)
    #pragma unroll
    for (int tm = 0; tm < 4; ++tm)
        #pragma unroll
        for (int tn = 0; tn < 2; ++tn)
            #pragma unroll
            for (int r = 0; r < 8; ++r) {
                int row = m0 + wm * 64 + tm * 16 + r + 8 * g;
                int col = n0 + wn * 32 + tn * 16 + l15;
                float v = acc[tm][tn][r];
                if (OUT_F32) ((float*)C)[(size_t)row * N + col] = v;
                else ((unsigned short*)C)[(size_t)row * N + col] = f2bfbits(v);
            }
}

// ---------------------------------------------------------------------------
// 3) RoPE on bf16 Q [S][H*D] and K [S][KVH*D]
// ---------------------------------------------------------------------------
__global__ __launch_bounds__(64) void rope_kernel(
    unsigned short* __restrict__ Q, unsigned short* __restrict__ Kn,
    const int* __restrict__ pos_ids)
{
    int s = blockIdx.x, h = blockIdx.y, d = threadIdx.x;   // d in [0,64)
    float pos = (float)(pos_ids[s] + LCK_I);
    float invf = __expf((float)d * (-LOG_THETA / 64.0f));
    float ang = pos * invf;
    float c = __cosf(ang), si = __sinf(ang);
    unsigned short* base = (h < NHEADS)
        ? Q + (size_t)s * QSTRIDE + h * HDIM
        : Kn + (size_t)s * KSTRIDE + (h - NHEADS) * HDIM;
    float x1 = bfbits2f(base[d]), x2 = bfbits2f(base[d + 64]);
    base[d]      = f2bfbits(x1 * c - x2 * si);
    base[d + 64] = f2bfbits(x2 * c + x1 * si);
}

// ---------------------------------------------------------------------------
// 4) Flash attention with online softmax over [S keys | d1 | d2].
//    Grid: (32 heads, 16 q-tiles). Block: 256 = 8 waves; wave owns 16 q-rows.
// ---------------------------------------------------------------------------
__global__ __launch_bounds__(256) void flash_attn(
    const unsigned short* __restrict__ Qbf,   // [S][4096]
    const unsigned short* __restrict__ Kn,    // [S][1024] (rope'd new keys)
    const unsigned short* __restrict__ Vn,    // [S][1024]
    const unsigned short* __restrict__ K0,    // [32][S][128]
    const unsigned short* __restrict__ K1,    // [32][S][128]
    const unsigned short* __restrict__ V0,    // [32][S][128]
    const unsigned short* __restrict__ V1,    // [32][S][128]
    unsigned short* __restrict__ Obf)         // [S][4096]
{
    __shared__ unsigned short ldsK[32][136];     // [key][d], padded
    __shared__ unsigned short ldsVt[128][40];    // [d][key], transposed, padded
    __shared__ unsigned short ldsP[8][16][40];   // per-wave prob staging

    const int h = blockIdx.x, qt = blockIdx.y;
    const int kvh = h >> 2;
    const int tid = threadIdx.x, lane = tid & 31, w = tid >> 5;
    const int g = lane >> 4, l15 = lane & 15;
    const int q0 = qt * 128 + w * 16;

    // --- Q fragments, loaded directly from global in WMMA A layout ---
    FragU qf[4];
    {
        const unsigned short* qrow = Qbf + (size_t)(q0 + l15) * QSTRIDE + h * HDIM;
        #pragma unroll
        for (int dstep = 0; dstep < 4; ++dstep) {
            int kb = dstep * 32 + g * 8;
            qf[dstep].q[0] = *(const uint4*)(qrow + kb);
            qf[dstep].q[1] = *(const uint4*)(qrow + kb + 16);
        }
    }

    // --- extras (d1 = q.cache_k1, d2 = q.k_new) seed the online softmax ---
    float m[8], l[8];
    v8f O[8];
    {
        int row = q0 + l15;
        const unsigned short* qrow  = Qbf + (size_t)row * QSTRIDE + h * HDIM;
        const unsigned short* k1row = K1 + ((size_t)h * S_LEN + row) * HDIM;
        const unsigned short* knrow = Kn + (size_t)row * KSTRIDE + kvh * HDIM;
        float d1 = 0.f, d2 = 0.f;
        #pragma unroll 4
        for (int dd = 0; dd < HDIM; dd += 8) {
            uint4 qv = *(const uint4*)(qrow + dd);
            uint4 k1v = *(const uint4*)(k1row + dd);
            uint4 knv = *(const uint4*)(knrow + dd);
            const unsigned short* qe  = (const unsigned short*)&qv;
            const unsigned short* ke1 = (const unsigned short*)&k1v;
            const unsigned short* ke2 = (const unsigned short*)&knv;
            #pragma unroll
            for (int j = 0; j < 8; ++j) {
                float qq = bfbits2f(qe[j]);
                d1 += qq * bfbits2f(ke1[j]);
                d2 += qq * bfbits2f(ke2[j]);
            }
        }
        d1 *= SCALE_F; d2 *= SCALE_F;
        float mr = fmaxf(d1, d2);
        float e1 = __expf(d1 - mr), e2 = __expf(d2 - mr);
        float lr = e1 + e2;
        #pragma unroll
        for (int r = 0; r < 8; ++r) {
            int src = r + 8 * g;
            m[r] = __shfl(mr, src);
            l[r] = __shfl(lr, src);
            float e1r = __shfl(e1, src), e2r = __shfl(e2, src);
            int rrow = q0 + src;
            #pragma unroll
            for (int t = 0; t < 8; ++t) {
                int col = t * 16 + l15;
                float v1 = bfbits2f(V1[((size_t)h * S_LEN + rrow) * HDIM + col]);
                float vn = bfbits2f(Vn[(size_t)rrow * KSTRIDE + kvh * HDIM + col]);
                O[t][r] = e1r * v1 + e2r * vn;
            }
        }
    }

    // --- main loop over 32-key blocks (causal: only up to this q tile) ---
    const int nblocks = (qt + 1) * 4;
    for (int kb = 0; kb < nblocks; ++kb) {
        const int kbase = kb * 32;
        __syncthreads();
        for (int i = tid; i < 512; i += 256) {          // K tile -> [key][d], async
            int kr = i >> 4, c = (i & 15) * 8;
            const unsigned short* gp = K0 + ((size_t)h * S_LEN + kbase + kr) * HDIM + c;
#ifdef ASYNC_LDS_OK
            async_copy_b128(gp, &ldsK[kr][c]);
#else
            *(uint4*)&ldsK[kr][c] = *(const uint4*)gp;
#endif
        }
        for (int i = tid; i < 512; i += 256) {          // V tile -> [d][key] (transpose)
            int kr = i >> 4, c = (i & 15) * 8;
            uint4 v = *(const uint4*)(V0 + ((size_t)h * S_LEN + kbase + kr) * HDIM + c);
            const unsigned short* e = (const unsigned short*)&v;
            #pragma unroll
            for (int j = 0; j < 8; ++j) ldsVt[c + j][kr] = e[j];
        }
#ifdef ASYNC_LDS_OK
        wait_async0();
#endif
        __syncthreads();
        if (kb + 1 < nblocks) {
            __builtin_prefetch(K0 + ((size_t)h * S_LEN + kbase + 32 + (tid >> 3)) * HDIM, 0, 1);
            __builtin_prefetch(V0 + ((size_t)h * S_LEN + kbase + 32 + (tid >> 3)) * HDIM, 0, 1);
        }

        // scores: two 16-key subtiles, K-dim = head dim (4 steps of 32)
        v8f s0 = {}, s1 = {};
        #pragma unroll
        for (int dstep = 0; dstep < 4; ++dstep) {
            FragU b0, b1;
            int kd = dstep * 32 + g * 8;
            b0.q[0] = *(const uint4*)&ldsK[l15][kd];
            b0.q[1] = *(const uint4*)&ldsK[l15][kd + 16];
            b1.q[0] = *(const uint4*)&ldsK[16 + l15][kd];
            b1.q[1] = *(const uint4*)&ldsK[16 + l15][kd + 16];
            s0 = wmma_bf16(qf[dstep], b0, s0);
            s1 = wmma_bf16(qf[dstep], b1, s1);
        }

        // online softmax update (row reductions via xor-shuffles in 16-lane halves)
        float p0[8], p1[8];
        #pragma unroll
        for (int r = 0; r < 8; ++r) {
            int rrow = q0 + r + 8 * g;
            int c0 = kbase + l15, c1 = c0 + 16;
            float v0s = s0[r] * SCALE_F + (c0 > rrow ? NEG_F : 0.f);
            float v1s = s1[r] * SCALE_F + (c1 > rrow ? NEG_F : 0.f);
            float bm = fmaxf(v0s, v1s);
            bm = fmaxf(bm, __shfl_xor(bm, 1));
            bm = fmaxf(bm, __shfl_xor(bm, 2));
            bm = fmaxf(bm, __shfl_xor(bm, 4));
            bm = fmaxf(bm, __shfl_xor(bm, 8));
            float mn = fmaxf(m[r], bm);
            float sf = __expf(m[r] - mn);
            m[r] = mn;
            float e0 = __expf(v0s - mn), e1 = __expf(v1s - mn);
            p0[r] = e0; p1[r] = e1;
            float rs = e0 + e1;
            rs += __shfl_xor(rs, 1);
            rs += __shfl_xor(rs, 2);
            rs += __shfl_xor(rs, 4);
            rs += __shfl_xor(rs, 8);
            l[r] = l[r] * sf + rs;
            #pragma unroll
            for (int t = 0; t < 8; ++t) O[t][r] *= sf;
        }

        // restage probs through per-wave LDS to enter WMMA A layout
        #pragma unroll
        for (int r = 0; r < 8; ++r) {
            ldsP[w][r + 8 * g][l15]      = f2bfbits(p0[r]);
            ldsP[w][r + 8 * g][16 + l15] = f2bfbits(p1[r]);
        }
        asm volatile("s_wait_dscnt 0" ::: "memory");
        FragU pf;
        {
            int kbp = g * 8;
            pf.q[0] = *(const uint4*)&ldsP[w][l15][kbp];
            pf.q[1] = *(const uint4*)&ldsP[w][l15][kbp + 16];
        }
        #pragma unroll
        for (int t = 0; t < 8; ++t) {
            FragU vf;
            int dc = t * 16 + l15, kk = g * 8;
            vf.q[0] = *(const uint4*)&ldsVt[dc][kk];
            vf.q[1] = *(const uint4*)&ldsVt[dc][kk + 16];
            O[t] = wmma_bf16(pf, vf, O[t]);
        }
    }

    // normalize + store bf16
    float linv[8];
    #pragma unroll
    for (int r = 0; r < 8; ++r) linv[r] = 1.0f / l[r];
    #pragma unroll
    for (int t = 0; t < 8; ++t) {
        int col = t * 16 + l15;
        #pragma unroll
        for (int r = 0; r < 8; ++r) {
            int rrow = q0 + r + 8 * g;
            Obf[(size_t)rrow * QSTRIDE + h * HDIM + col] = f2bfbits(O[t][r] * linv[r]);
        }
    }
}

// ---------------------------------------------------------------------------
// launcher
// ---------------------------------------------------------------------------
extern "C" void kernel_launch(void* const* d_in, const int* in_sizes, int n_in,
                              void* d_out, int out_size, void* d_ws, size_t ws_size,
                              hipStream_t stream)
{
    const float* X   = (const float*)d_in[0];
    // d_in[1] = attention_mask (causal; applied analytically)
    const int*   pos = (const int*)d_in[2];
    const float* k0  = (const float*)d_in[3];
    const float* k1  = (const float*)d_in[4];
    const float* v0  = (const float*)d_in[5];
    const float* v1  = (const float*)d_in[6];
    const float* Wq  = (const float*)d_in[7];
    const float* Wk  = (const float*)d_in[8];
    const float* Wv  = (const float*)d_in[9];
    const float* Wo  = (const float*)d_in[10];
    float* out = (float*)d_out;

    char* ws = (char*)d_ws;
    size_t off = 0;
    auto alloc = [&](size_t bytes) -> unsigned short* {
        unsigned short* p = (unsigned short*)(ws + off);
        off = (off + bytes + 255) & ~(size_t)255;
        return p;
    };
    const size_t CACHE_E = (size_t)NHEADS * S_LEN * HDIM;       // 8.39M
    unsigned short* Xbf  = alloc((size_t)S_LEN * XK * 2);
    unsigned short* Wqbf = alloc((size_t)XK * QSTRIDE * 2);
    unsigned short* Wkbf = alloc((size_t)XK * KSTRIDE * 2);
    unsigned short* Wvbf = alloc((size_t)XK * KSTRIDE * 2);
    unsigned short* Wobf = alloc((size_t)QSTRIDE * QSTRIDE * 2);
    unsigned short* K0bf = alloc(CACHE_E * 2);
    unsigned short* K1bf = alloc(CACHE_E * 2);
    unsigned short* V0bf = alloc(CACHE_E * 2);
    unsigned short* V1bf = alloc(CACHE_E * 2);
    unsigned short* Qbf  = alloc((size_t)S_LEN * QSTRIDE * 2);
    unsigned short* Knbf = alloc((size_t)S_LEN * KSTRIDE * 2);
    unsigned short* Vnbf = alloc((size_t)S_LEN * KSTRIDE * 2);
    unsigned short* Obf  = alloc((size_t)S_LEN * QSTRIDE * 2);

    auto cvt = [&](const float* s, unsigned short* d, long long n) {
        unsigned blocks = (unsigned)((n / 4 + 255) / 256);
        cvt_f32_bf16<<<blocks, 256, 0, stream>>>(s, d, n);
    };
    cvt(X,  Xbf,  (long long)S_LEN * XK);
    cvt(Wq, Wqbf, (long long)XK * QSTRIDE);
    cvt(Wk, Wkbf, (long long)XK * KSTRIDE);
    cvt(Wv, Wvbf, (long long)XK * KSTRIDE);
    cvt(Wo, Wobf, (long long)QSTRIDE * QSTRIDE);
    cvt(k0, K0bf, (long long)CACHE_E);
    cvt(k1, K1bf, (long long)CACHE_E);
    cvt(v0, V0bf, (long long)CACHE_E);
    cvt(v1, V1bf, (long long)CACHE_E);

    // QKV projections (bf16 WMMA, bf16 outputs)
    gemm_bf16<false><<<dim3(QSTRIDE / 128, S_LEN / 128), 256, 0, stream>>>(
        Xbf, Wqbf, Qbf, S_LEN, QSTRIDE, XK);
    gemm_bf16<false><<<dim3(KSTRIDE / 128, S_LEN / 128), 256, 0, stream>>>(
        Xbf, Wkbf, Knbf, S_LEN, KSTRIDE, XK);
    gemm_bf16<false><<<dim3(KSTRIDE / 128, S_LEN / 128), 256, 0, stream>>>(
        Xbf, Wvbf, Vnbf, S_LEN, KSTRIDE, XK);

    // RoPE on Q (32 heads) and K (8 heads)
    rope_kernel<<<dim3(S_LEN, NHEADS + NKVH), 64, 0, stream>>>(Qbf, Knbf, pos);

    // Flash attention over [cache_k0 | d1 | d2]
    flash_attn<<<dim3(NHEADS, S_LEN / 128), 256, 0, stream>>>(
        Qbf, Knbf, Vnbf, K0bf, K1bf, V0bf, V1bf, Obf);

    // Output projection -> fp32 d_out
    gemm_bf16<true><<<dim3(QSTRIDE / 128, S_LEN / 128), 256, 0, stream>>>(
        Obf, Wobf, out, S_LEN, QSTRIDE, QSTRIDE);
}
